// BPF_Attention_1675037245749
// MI455X (gfx1250) — matrix operand order repaired and checked by reading, not compiled, not run on hardware
//
#include <hip/hip_runtime.h>
#include <hip/hip_bf16.h>

// ---------------------------------------------------------------------------
// MI455X (gfx1250) masked multi-head attention block, bf16 WMMA pipeline.
// B=4, N=2048, C=768, H=12, D=64.  wave32, V_WMMA_F32_16X16X32_BF16,
// GLOBAL_LOAD_ASYNC_TO_LDS_B128 for bf16 tile staging.
// ---------------------------------------------------------------------------

typedef __bf16 bf16;
typedef __attribute__((ext_vector_type(16))) __bf16 v16bf;
typedef __attribute__((ext_vector_type(8)))  float  v8f;

#define BATCH   4
#define SEQ     2048
#define DIM     768
#define HEADS   12
#define HDIM    64
#define DIM3    2304
#define MROWS   (BATCH * SEQ)       // 8192
#define SCALE   0.125f              // 64^-0.5

static __device__ __forceinline__ bf16 f2bf(float f) {
    unsigned u = __builtin_bit_cast(unsigned, f);
    unsigned r = (u + 0x7FFFu + ((u >> 16) & 1u)) >> 16;
    unsigned short s = (unsigned short)r;
    return __builtin_bit_cast(bf16, s);
}

union FragU { v16bf v; int4 q[2]; };

// A-matrix fragment (16x32 bf16), src row-major with stride ld (elements).
// lanes 0-15: row m=lane, e0..7 -> K=k0+0..7,  e8..15 -> K=k0+16..23
// lanes16-31: row m=lane-16, e0..7 -> K=k0+8..15, e8..15 -> K=k0+24..31
static __device__ __forceinline__ v16bf load_frag_a(const bf16* src, int ld,
                                                    int lane, int k0) {
    int m = lane & 15, h = lane >> 4;
    FragU f;
    f.q[0] = *(const int4*)(src + (size_t)m * ld + k0 + h * 8);
    f.q[1] = *(const int4*)(src + (size_t)m * ld + k0 + 16 + h * 8);
    return f.v;
}

// B-matrix fragment (32x16, K x N) from N-major storage (element (n,k) at
// n*ld + k).  lane column n = n0 + lane%16, e0..7 -> K = k0 + half*8 + e,
// e8..15 -> K = k0 + 16 + half*8 + (e-8).
static __device__ __forceinline__ v16bf load_frag_b(const bf16* srcT, int ld,
                                                    int lane, int n0, int k0) {
    int n = lane & 15, h = lane >> 4;
    const bf16* p = srcT + (size_t)(n0 + n) * ld + k0 + h * 8;
    FragU f;
    f.q[0] = *(const int4*)(p);
    f.q[1] = *(const int4*)(p + 16);
    return f.v;
}

static __device__ __forceinline__ v8f wmma_bf16(v16bf a, v16bf b, v8f c) {
    return __builtin_amdgcn_wmma_f32_16x16x32_bf16(
        /*neg_a=*/false, a, /*neg_b=*/false, b,
        /*c_mod=*/(short)0, c, /*reuse_a=*/false, /*reuse_b=*/false);
}

// Async global -> LDS copy of 16 bytes per lane (CDNA5 ASYNCcnt path).
// VDST = LDS byte offset (low 32 bits of generic LDS pointer), VADDR = 64-bit
// global address, no SADDR.
static __device__ __forceinline__ void gld_async_b128(void* lds, const void* g) {
    unsigned ldsoff = (unsigned)(size_t)lds;
    asm volatile("global_load_async_to_lds_b128 %0, %1, off"
                 :: "v"(ldsoff), "v"(g) : "memory");
}
static __device__ __forceinline__ void wait_async0() {
    asm volatile("s_wait_asynccnt 0x0" ::: "memory");
}

// ---------------------------------------------------------------------------
// Weight transpose + fp32 -> bf16 convert:  wT[c*rows + r] = w[r*cols + c]
// ---------------------------------------------------------------------------
__global__ void k_transpose_w(const float* __restrict__ w, bf16* __restrict__ wT,
                              int rows, int cols) {
    int c = blockIdx.x * 16 + threadIdx.x;
    int r = blockIdx.y * 16 + threadIdx.y;
    if (r < rows && c < cols)
        wT[(size_t)c * rows + r] = f2bf(w[(size_t)r * cols + c]);
}

// ---------------------------------------------------------------------------
// QKV GEMM: [8192 x 768] (fp32 x -> bf16 in LDS) @ wqkvT -> scatter into
// q[B,H,N,D], k[B,H,N,D], vT[B,H,D,N] as bf16.
// Block: 256 threads = 8 waves; 128x64 tile; wave = 32x32 (4 WMMA / K-step).
// B tile staged with async global->LDS.
// ---------------------------------------------------------------------------
__global__ __launch_bounds__(256) void k_qkv_gemm(
    const float* __restrict__ x, const bf16* __restrict__ wqkvT,
    bf16* __restrict__ q, bf16* __restrict__ k, bf16* __restrict__ vT) {
    __shared__ __align__(16) bf16 sA[128 * 32];   // 8 KB
    __shared__ __align__(16) bf16 sB[64 * 32];    // 4 KB

    const int m0 = blockIdx.x * 128;
    const int n0 = blockIdx.y * 64;
    const int tid = threadIdx.x;
    const int lane = tid & 31, wave = tid >> 5;
    const int wr = wave & 3, wc = wave >> 2;          // 4 row x 2 col wave grid
    const int arow = tid >> 1, aseg = (tid & 1) * 16; // A: 2 thr/row, 16 f32
    const int brow = tid >> 2, bseg = (tid & 3) * 8;  // B: 4 thr/row, 8 bf16

    v8f acc[2][2] = {};

    for (int k0 = 0; k0 < DIM; k0 += 32) {
        // A tile: 128x32 fp32 -> bf16 through VGPRs (needs conversion)
        const float* xp = x + (size_t)(m0 + arow) * DIM + k0 + aseg;
        float4 f0 = *(const float4*)(xp);
        float4 f1 = *(const float4*)(xp + 4);
        float4 f2 = *(const float4*)(xp + 8);
        float4 f3 = *(const float4*)(xp + 12);
        if (k0 + 32 < DIM) __builtin_prefetch(xp + 32, 0, 0);
        bf16* ap = sA + arow * 32 + aseg;
        ap[0]  = f2bf(f0.x); ap[1]  = f2bf(f0.y); ap[2]  = f2bf(f0.z); ap[3]  = f2bf(f0.w);
        ap[4]  = f2bf(f1.x); ap[5]  = f2bf(f1.y); ap[6]  = f2bf(f1.z); ap[7]  = f2bf(f1.w);
        ap[8]  = f2bf(f2.x); ap[9]  = f2bf(f2.y); ap[10] = f2bf(f2.z); ap[11] = f2bf(f2.w);
        ap[12] = f2bf(f3.x); ap[13] = f2bf(f3.y); ap[14] = f2bf(f3.z); ap[15] = f2bf(f3.w);
        // B tile: 64x32 bf16, async global -> LDS (no VGPR round-trip)
        gld_async_b128(sB + brow * 32 + bseg,
                       wqkvT + (size_t)(n0 + brow) * DIM + k0 + bseg);
        wait_async0();
        __syncthreads();

        v16bf a0 = load_frag_a(sA + (wr * 32) * 32,      32, lane, 0);
        v16bf a1 = load_frag_a(sA + (wr * 32 + 16) * 32, 32, lane, 0);
        v16bf b0 = load_frag_b(sB, 32, lane, wc * 32, 0);
        v16bf b1 = load_frag_b(sB, 32, lane, wc * 32 + 16, 0);
        acc[0][0] = wmma_bf16(a0, b0, acc[0][0]);
        acc[0][1] = wmma_bf16(a0, b1, acc[0][1]);
        acc[1][0] = wmma_bf16(a1, b0, acc[1][0]);
        acc[1][1] = wmma_bf16(a1, b1, acc[1][1]);
        __syncthreads();
    }

    // Scatter epilogue: C/D layout row = j + 8*(lane/16), col = lane%16
    const int half = lane >> 4, nn = lane & 15;
#pragma unroll
    for (int ti = 0; ti < 2; ++ti) {
#pragma unroll
        for (int tj = 0; tj < 2; ++tj) {
            int c3 = n0 + wc * 32 + tj * 16 + nn;
            int sec = c3 / DIM, ch = c3 % DIM;
            int h = ch >> 6, d = ch & 63;
#pragma unroll
            for (int j = 0; j < 8; ++j) {
                int gr = m0 + wr * 32 + ti * 16 + j + half * 8;
                int b_ = gr >> 11, nq = gr & (SEQ - 1);
                size_t bh = (size_t)(b_ * HEADS + h);
                bf16 val = f2bf(acc[ti][tj][j]);
                if (sec == 0)      q[(bh * SEQ + nq) * HDIM + d] = val;
                else if (sec == 1) k[(bh * SEQ + nq) * HDIM + d] = val;
                else               vT[(bh * HDIM + d) * SEQ + nq] = val;
            }
        }
    }
}

// ---------------------------------------------------------------------------
// Flash-style masked attention. Block = 256 thr = 8 waves; one block handles
// (head bh, 128 query rows). Wave owns 16 query rows; Q A-frags in registers,
// K/V B-frags direct from global (L2-resident), P staged via wave-private LDS.
// ---------------------------------------------------------------------------
__global__ __launch_bounds__(256) void k_attn(
    const bf16* __restrict__ q, const bf16* __restrict__ k,
    const bf16* __restrict__ vT, const unsigned char* __restrict__ mask,
    bf16* __restrict__ attn_out) {
    __shared__ __align__(16) bf16 sP[8][16 * 64];   // per-wave P buffer (16KB)

    const int bh = blockIdx.x;                 // 0..47
    const int m0 = blockIdx.y * 128;           // query row tile
    const int b  = bh / HEADS, h = bh % HEADS;
    const int tid = threadIdx.x, lane = tid & 31, wave = tid >> 5;
    const int half = lane >> 4, nn = lane & 15;

    const bf16* qb = q  + (size_t)bh * SEQ * HDIM;
    const bf16* kb = k  + (size_t)bh * SEQ * HDIM;
    const bf16* vb = vT + (size_t)bh * HDIM * SEQ;

    const int qrow0 = m0 + wave * 16;
    // Q fragments (K = d dimension, 0..63) held in registers for whole kernel
    v16bf qa0 = load_frag_a(qb + (size_t)qrow0 * HDIM, HDIM, lane, 0);
    v16bf qa1 = load_frag_a(qb + (size_t)qrow0 * HDIM, HDIM, lane, 32);

    float mrun[8], lrun[8];
    v8f o[4] = {};
#pragma unroll
    for (int j = 0; j < 8; ++j) { mrun[j] = -3.0e38f; lrun[j] = 0.0f; }

    for (int kt = 0; kt < SEQ; kt += 64) {
        // ---- S = Q @ K^T  (4 tiles of 16 keys, K-dim = 64 head dims) ----
        v8f s[4] = {};
#pragma unroll
        for (int t = 0; t < 4; ++t) {
            v16bf b0 = load_frag_b(kb, HDIM, lane, kt + t * 16, 0);
            v16bf b1 = load_frag_b(kb, HDIM, lane, kt + t * 16, 32);
            s[t] = wmma_bf16(qa0, b0, s[t]);
            s[t] = wmma_bf16(qa1, b1, s[t]);
        }

        // ---- online softmax over the 64-key slab ----
        bf16* pbuf = sP[wave];
#pragma unroll
        for (int j = 0; j < 8; ++j) {
            int grow = qrow0 + j + half * 8;
            const unsigned char* mrow = mask + (size_t)grow * SEQ + kt;
            float sv[4];
            float mx = -3.0e38f;
#pragma unroll
            for (int t = 0; t < 4; ++t) {
                float v = s[t][j] * SCALE;
                if (mrow[t * 16 + nn]) v = -3.0e38f;   // masked_fill(-inf)
                sv[t] = v;
                mx = fmaxf(mx, v);
            }
#pragma unroll
            for (int off = 1; off < 16; off <<= 1)
                mx = fmaxf(mx, __shfl_xor(mx, off, 32));
            float mnew  = fmaxf(mrun[j], mx);
            float alpha = __expf(mrun[j] - mnew);
            float sum = 0.0f;
#pragma unroll
            for (int t = 0; t < 4; ++t) {
                float p = __expf(sv[t] - mnew);
                sum += p;
                pbuf[(j + half * 8) * 64 + t * 16 + nn] = f2bf(p);
            }
#pragma unroll
            for (int off = 1; off < 16; off <<= 1)
                sum += __shfl_xor(sum, off, 32);
            lrun[j] = lrun[j] * alpha + sum;
            mrun[j] = mnew;
#pragma unroll
            for (int t = 0; t < 4; ++t) o[t][j] *= alpha;
        }

        // ---- O += P @ V  (A = P 16x64 from LDS, B = vT slab) ----
        v16bf pa0 = load_frag_a(pbuf, 64, lane, 0);
        v16bf pa1 = load_frag_a(pbuf, 64, lane, 32);
#pragma unroll
        for (int td = 0; td < 4; ++td) {
            v16bf bv0 = load_frag_b(vb, SEQ, lane, td * 16, kt);
            v16bf bv1 = load_frag_b(vb, SEQ, lane, td * 16, kt + 32);
            o[td] = wmma_bf16(pa0, bv0, o[td]);
            o[td] = wmma_bf16(pa1, bv1, o[td]);
        }
    }

    // ---- normalize and emit attn_out[(b*N+nq)*C + h*64 + d] (bf16) ----
#pragma unroll
    for (int j = 0; j < 8; ++j) {
        float inv = lrun[j] > 0.0f ? 1.0f / lrun[j] : 0.0f;
        int grow = qrow0 + j + half * 8;
        size_t orow = ((size_t)b * SEQ + grow) * DIM + h * HDIM;
#pragma unroll
        for (int td = 0; td < 4; ++td)
            attn_out[orow + td * 16 + nn] = f2bf(o[td][j] * inv);
    }
}

// ---------------------------------------------------------------------------
// Projection GEMM: out = attn_out(bf16 [8192x768]) @ wprojT + b_proj, fp32 out
// Both tiles staged with async global->LDS (pure bf16 copies).
// ---------------------------------------------------------------------------
__global__ __launch_bounds__(256) void k_proj_gemm(
    const bf16* __restrict__ A, const bf16* __restrict__ wprojT,
    const float* __restrict__ bias, float* __restrict__ out) {
    __shared__ __align__(16) bf16 sA[128 * 32];   // 8 KB
    __shared__ __align__(16) bf16 sB[64 * 32];    // 4 KB

    const int m0 = blockIdx.x * 128;
    const int n0 = blockIdx.y * 64;
    const int tid = threadIdx.x;
    const int lane = tid & 31, wave = tid >> 5;
    const int wr = wave & 3, wc = wave >> 2;
    const int arow = tid >> 1, aseg = (tid & 1) * 16;  // A: 2 thr/row, 16 bf16
    const int brow = tid >> 2, bseg = (tid & 3) * 8;   // B: 4 thr/row, 8 bf16

    v8f acc[2][2] = {};

    for (int k0 = 0; k0 < DIM; k0 += 32) {
        const bf16* agp = A + (size_t)(m0 + arow) * DIM + k0 + aseg;
        gld_async_b128(sA + arow * 32 + aseg,     agp);
        gld_async_b128(sA + arow * 32 + aseg + 8, agp + 8);
        gld_async_b128(sB + brow * 32 + bseg,
                       wprojT + (size_t)(n0 + brow) * DIM + k0 + bseg);
        if (k0 + 32 < DIM) __builtin_prefetch(agp + 32, 0, 0);
        wait_async0();
        __syncthreads();

        v16bf a0 = load_frag_a(sA + (wr * 32) * 32,      32, lane, 0);
        v16bf a1 = load_frag_a(sA + (wr * 32 + 16) * 32, 32, lane, 0);
        v16bf b0 = load_frag_b(sB, 32, lane, wc * 32, 0);
        v16bf b1 = load_frag_b(sB, 32, lane, wc * 32 + 16, 0);
        acc[0][0] = wmma_bf16(a0, b0, acc[0][0]);
        acc[0][1] = wmma_bf16(a0, b1, acc[0][1]);
        acc[1][0] = wmma_bf16(a1, b0, acc[1][0]);
        acc[1][1] = wmma_bf16(a1, b1, acc[1][1]);
        __syncthreads();
    }

    const int half = lane >> 4, nn = lane & 15;
#pragma unroll
    for (int ti = 0; ti < 2; ++ti) {
#pragma unroll
        for (int tj = 0; tj < 2; ++tj) {
            int c = n0 + wc * 32 + tj * 16 + nn;
            float bv = bias[c];
#pragma unroll
            for (int j = 0; j < 8; ++j) {
                int gr = m0 + wr * 32 + ti * 16 + j + half * 8;
                out[(size_t)gr * DIM + c] = acc[ti][tj][j] + bv;
            }
        }
    }
}

// ---------------------------------------------------------------------------
extern "C" void kernel_launch(void* const* d_in, const int* in_sizes, int n_in,
                              void* d_out, int out_size, void* d_ws, size_t ws_size,
                              hipStream_t stream) {
    const float*         x      = (const float*)d_in[0];
    const unsigned char* mask   = (const unsigned char*)d_in[1];  // bool [N,N]
    const float*         w_qkv  = (const float*)d_in[2];          // [C, 3C]
    const float*         w_proj = (const float*)d_in[3];          // [C, C]
    const float*         b_proj = (const float*)d_in[4];          // [C]
    float*               out    = (float*)d_out;                  // [B,N,C]
    char*                ws     = (char*)d_ws;

    const size_t QB = (size_t)BATCH * HEADS * SEQ * HDIM * sizeof(bf16); // 12.58MB
    bf16* q      = (bf16*)(ws);
    bf16* kk     = (bf16*)(ws + QB);
    bf16* vT     = (bf16*)(ws + 2 * QB);
    bf16* ao     = (bf16*)(ws + 3 * QB);
    bf16* wqkvT  = (bf16*)(ws + 4 * QB);
    bf16* wprojT = (bf16*)(ws + 4 * QB + (size_t)DIM3 * DIM * sizeof(bf16));

    dim3 b16(16, 16);
    k_transpose_w<<<dim3(DIM3 / 16, DIM / 16), b16, 0, stream>>>(w_qkv, wqkvT, DIM, DIM3);
    k_transpose_w<<<dim3(DIM / 16, DIM / 16), b16, 0, stream>>>(w_proj, wprojT, DIM, DIM);

    k_qkv_gemm<<<dim3(MROWS / 128, DIM3 / 64), 256, 0, stream>>>(x, wqkvT, q, kk, vT);

    k_attn<<<dim3(BATCH * HEADS, SEQ / 128), 256, 0, stream>>>(q, kk, vT, mask, ao);

    k_proj_gemm<<<dim3(MROWS / 128, DIM / 64), 256, 0, stream>>>(ao, wprojT, b_proj, out);
}
